// GATRec_62199716380752
// MI455X (gfx1250) — matrix operand order repaired and checked by reading, not compiled, not run on hardware
//
#include <hip/hip_runtime.h>

// ---------------- problem constants (from reference) ----------------
#define N_USERS   100000
#define N_ITEMS   20000
#define N_NODES   120000
#define N_GENRES  20
#define N_TAGS    1128
#define EMB       64
#define DM        192      // D_MOVIE = 64 + 64 + 64
#define HID       256
#define H1        8        // heads layer 1
#define C1V       32       // channels/head layer 1
#define MOVX_LD   1148     // N_GENRES + N_TAGS
#define KP_G      32       // padded K for genre features
#define KP_T      1152     // padded K for tag features (1128 -> 36*32)
#define NEG_SLOPE 0.2f

typedef __attribute__((ext_vector_type(8)))  __bf16 bf16x8;
typedef __attribute__((ext_vector_type(16))) __bf16 bf16x16;
typedef __attribute__((ext_vector_type(8)))  float  f32x8;

// ---------------- helpers ----------------
__device__ __forceinline__ __bf16 to_bf16(float f) {
    unsigned u = __float_as_uint(f);
    u += 0x7fffu + ((u >> 16) & 1u);           // round-to-nearest-even
    unsigned short h = (unsigned short)(u >> 16);
    __bf16 r; __builtin_memcpy(&r, &h, 2); return r;
}
__device__ __forceinline__ float bf_to_f(__bf16 b) {
    unsigned short h; __builtin_memcpy(&h, &b, 2);
    return __uint_as_float(((unsigned)h) << 16);
}
// order-preserving float<->uint key for atomic segment-max
__device__ __forceinline__ unsigned f2key(float f) {
    unsigned u = __float_as_uint(f);
    return (u & 0x80000000u) ? ~u : (u | 0x80000000u);
}
__device__ __forceinline__ float key2f(unsigned k) {
    unsigned u = (k & 0x80000000u) ? (k & 0x7fffffffu) : ~k;
    return __uint_as_float(u);
}

// ---------------- generic fill ----------------
__global__ void fill_k(unsigned* __restrict__ p, unsigned v, long n) {
    long i = blockIdx.x * 256L + threadIdx.x;
    if (i < n) p[i] = v;
}

// ---------------- weight transpose + bf16 (Wt[n*Kp+k] = W[k*N+n], zero-pad K) ----------------
__global__ void cvtw_k(const float* __restrict__ W, __bf16* __restrict__ Wt,
                       int K, int N, int Kp) {
    long idx = blockIdx.x * 256L + threadIdx.x;
    long total = (long)N * Kp;
    if (idx >= total) return;
    int n = (int)(idx / Kp), k = (int)(idx % Kp);
    Wt[idx] = (k < K) ? to_bf16(W[(long)k * N + n]) : to_bf16(0.f);
}

// ---------------- attention-vector folding: out[k*H+h] = sum_c W[k, h*C+c]*a[h*C+c] ----------------
__global__ void fold_k(const float* __restrict__ W, const float* __restrict__ a,
                       float* __restrict__ out, int K, int N, int H, int C) {
    int idx = blockIdx.x * 256 + threadIdx.x;
    if (idx >= K * H) return;
    int k = idx / H, h = idx % H;
    float s = 0.f;
    for (int c = 0; c < C; ++c) s += W[(long)k * N + h * C + c] * a[h * C + c];
    out[idx] = s;
}

// ---------------- movie feature slices -> padded bf16 ----------------
__global__ void featcvt_k(const float* __restrict__ mov,
                          __bf16* __restrict__ gfb, __bf16* __restrict__ tfb) {
    long idx = blockIdx.x * 256L + threadIdx.x;
    long total = (long)N_ITEMS * (KP_G + KP_T);
    if (idx >= total) return;
    int m = (int)(idx / (KP_G + KP_T));
    int j = (int)(idx % (KP_G + KP_T));
    if (j < KP_G) {
        gfb[(long)m * KP_G + j] =
            (j < N_GENRES) ? to_bf16(mov[(long)m * MOVX_LD + j]) : to_bf16(0.f);
    } else {
        int jj = j - KP_G;
        tfb[(long)m * KP_T + jj] =
            (jj < N_TAGS) ? to_bf16(mov[(long)m * MOVX_LD + N_GENRES + jj]) : to_bf16(0.f);
    }
}

// ---------------- f32 -> bf16 ----------------
__global__ void f2b_k(const float* __restrict__ in, __bf16* __restrict__ out, long n) {
    long i = blockIdx.x * 256L + threadIdx.x;
    if (i < n) out[i] = to_bf16(in[i]);
}

// ---------------- WMMA bf16 GEMM: C[M,N] = A[M,Kp] * Bt[N,Kp]^T (+bias, opt relu) ----------------
// One wave per 16-row M tile and NT 16-col N tiles; f32 accumulate.
// Per k-step: batch-load A + all NT B fragments first (one load clause, partial
// loadcnt waits), then issue NT back-to-back WMMAs.
template <int NT>
__global__ void __launch_bounds__(256) wmma_gemm_k(
    const __bf16* __restrict__ A, const __bf16* __restrict__ Bt,
    float* __restrict__ C, const float* __restrict__ bias,
    int M, int N, int Kp, int doRelu)
{
    const int lane = threadIdx.x & 31;
    const int wid  = blockIdx.x * (blockDim.x >> 5) + (threadIdx.x >> 5);
    const int MT = M >> 4;
    const int NB = N / (NT * 16);
    if (wid >= MT * NB) return;              // wave-uniform: EXEC stays all-ones for WMMA
    const int mt = wid % MT;
    const int nb = wid / MT;

    const int lm   = lane & 15;
    const int lhi  = lane >> 4;              // 0 or 1
    const int aoff = lhi << 4;               // A: hi half-wave holds K+16 block
    const int boff = lhi << 4;               // B: hi half-wave holds K+16 block

    f32x8 acc[NT];
#pragma unroll
    for (int i = 0; i < NT; ++i) {
        f32x8 z = {0.f, 0.f, 0.f, 0.f, 0.f, 0.f, 0.f, 0.f};
        acc[i] = z;
    }

    const __bf16* arow = A + (long)(mt * 16 + lm) * Kp + aoff;
    const __bf16* brow0 = Bt + (long)(nb * NT * 16 + lm) * Kp + boff;

    for (int k0 = 0; k0 < Kp; k0 += 32) {
        // ---- issue ALL loads for this k-step first ----
        bf16x8 a_lo = *(const bf16x8*)(arow + k0);        // K {0..7} or {16..23}
        bf16x8 a_hi = *(const bf16x8*)(arow + k0 + 8);    // K {8..15} or {24..31}
        bf16x16 bv[NT];
#pragma unroll
        for (int nt = 0; nt < NT; ++nt) {
            const __bf16* brow = brow0 + (long)(nt * 16) * Kp + k0;
            bf16x8 b_lo = *(const bf16x8*)(brow);         // 16 contiguous K per lane
            bf16x8 b_hi = *(const bf16x8*)(brow + 8);
            bv[nt] = __builtin_shufflevector(b_lo, b_hi,
                0,1,2,3,4,5,6,7,8,9,10,11,12,13,14,15);
        }
        bf16x16 af = __builtin_shufflevector(a_lo, a_hi,
            0,1,2,3,4,5,6,7,8,9,10,11,12,13,14,15);
        // ---- then a WMMA burst ----
#pragma unroll
        for (int nt = 0; nt < NT; ++nt) {
            acc[nt] = __builtin_amdgcn_wmma_f32_16x16x32_bf16(
                false, af, false, bv[nt], (short)0, acc[nt], false, false);
        }
    }

    // C/D layout: VGPR r -> M = r (+8 for hi half-wave), N = lane&15
    const int rbase = mt * 16 + (lhi << 3);
#pragma unroll
    for (int nt = 0; nt < NT; ++nt) {
        const int n = nb * NT * 16 + nt * 16 + lm;
        float bv2 = bias ? bias[n] : 0.f;
#pragma unroll
        for (int r = 0; r < 8; ++r) {
            float v = acc[nt][r] + bv2;
            if (doRelu) v = v > 0.f ? v : 0.f;
            C[(long)(rbase + r) * N + n] = v;
        }
    }
}

// ---------------- build x (bf16, 120000 x 192) ----------------
__global__ void build_x_k(const float* __restrict__ uemb, const float* __restrict__ iemb,
                          const float* __restrict__ g, const float* __restrict__ t,
                          __bf16* __restrict__ xb) {
    long idx = blockIdx.x * 256L + threadIdx.x;
    long total = (long)N_NODES * DM;
    if (idx >= total) return;
    int n = (int)(idx / DM), k = (int)(idx % DM);
    float v;
    if (n < N_USERS) {
        v = (k < EMB) ? uemb[(long)n * EMB + k] : 0.f;
    } else {
        int i = n - N_USERS;
        if (k < EMB)            v = iemb[(long)i * EMB + k];
        else if (k < 2 * EMB)   v = g[(long)i * EMB + (k - EMB)];
        else                    v = t[(long)i * EMB + (k - 2 * EMB)];
    }
    xb[idx] = to_bf16(v);
}

// ---------------- al_s / al_d via folded vectors: als[n,h] = sum_k x[n,k]*Wsa[k,h] ----------------
template <int H>
__global__ void __launch_bounds__(256) al_k(const __bf16* __restrict__ X,
    const float* __restrict__ Wsa, const float* __restrict__ Wda,
    float* __restrict__ als, float* __restrict__ ald, int K)
{
    __shared__ float sS[1536];
    __shared__ float sD[1536];
    for (int i = threadIdx.x; i < K * H; i += blockDim.x) { sS[i] = Wsa[i]; sD[i] = Wda[i]; }
    __syncthreads();
    int n = blockIdx.x * 256 + threadIdx.x;
    if (n >= N_NODES) return;
    float s[H], d[H];
#pragma unroll
    for (int h = 0; h < H; ++h) { s[h] = 0.f; d[h] = 0.f; }
    const __bf16* xr = X + (long)n * K;
    for (int k8 = 0; k8 < K; k8 += 8) {
        bf16x8 v = *(const bf16x8*)(xr + k8);
#pragma unroll
        for (int q = 0; q < 8; ++q) {
            float xv = bf_to_f(v[q]);
            int k = k8 + q;
#pragma unroll
            for (int h = 0; h < H; ++h) {
                s[h] += xv * sS[k * H + h];
                d[h] += xv * sD[k * H + h];
            }
        }
    }
#pragma unroll
    for (int h = 0; h < H; ++h) { als[(long)n * H + h] = s[h]; ald[(long)n * H + h] = d[h]; }
}

// ---------------- edge passes ----------------
__global__ void edge_max_k(const int* __restrict__ src, const int* __restrict__ dst,
                           const float* __restrict__ als, const float* __restrict__ ald,
                           unsigned* __restrict__ mmaxk, int E_, int hbits) {
    const int H = 1 << hbits;
    long tid = blockIdx.x * 256L + threadIdx.x;
    if (tid >= ((long)E_ << hbits)) return;
    int h = (int)(tid & (H - 1));
    int e = (int)(tid >> hbits);
    int s = src[e], d = dst[e];
    float x = als[(long)s * H + h] + ald[(long)d * H + h];
    x = x > 0.f ? x : NEG_SLOPE * x;
    atomicMax(&mmaxk[(long)d * H + h], f2key(x));
}

__global__ void edge_sum_k(const int* __restrict__ src, const int* __restrict__ dst,
                           const float* __restrict__ als, const float* __restrict__ ald,
                           const unsigned* __restrict__ mmaxk, float* __restrict__ ssum,
                           int E_, int hbits) {
    const int H = 1 << hbits;
    long tid = blockIdx.x * 256L + threadIdx.x;
    if (tid >= ((long)E_ << hbits)) return;
    int h = (int)(tid & (H - 1));
    int e = (int)(tid >> hbits);
    int s = src[e], d = dst[e];
    float x = als[(long)s * H + h] + ald[(long)d * H + h];
    x = x > 0.f ? x : NEG_SLOPE * x;
    float w = __expf(x - key2f(mmaxk[(long)d * H + h]));
    atomicAdd(&ssum[(long)d * H + h], w);
}

__global__ void __launch_bounds__(256) edge_aggr_k(
    const int* __restrict__ src, const int* __restrict__ dst,
    const float* __restrict__ als, const float* __restrict__ ald,
    const unsigned* __restrict__ mmaxk, const float* __restrict__ ssum,
    const float* __restrict__ hs, float* __restrict__ acc,
    int E_, int H, int cshift)
{
    int c = threadIdx.x;          // 0..255 output channel
    int h = c >> cshift;
    for (int e = blockIdx.x; e < E_; e += gridDim.x) {
        int s = src[e], d = dst[e];
        float x = als[(long)s * H + h] + ald[(long)d * H + h];
        x = x > 0.f ? x : NEG_SLOPE * x;
        float w = __expf(x - key2f(mmaxk[(long)d * H + h]));
        float alpha = w / (ssum[(long)d * H + h] + 1e-16f);
        atomicAdd(&acc[(long)d * HID + c], hs[(long)s * HID + c] * alpha);
    }
}

// ---------------- finalizers ----------------
__global__ void fin1_k(const float* __restrict__ acc, const float* __restrict__ b1,
                       __bf16* __restrict__ ob) {
    long idx = blockIdx.x * 256L + threadIdx.x;
    if (idx >= (long)N_NODES * HID) return;
    int c = (int)(idx & (HID - 1));
    float v = acc[idx] + b1[c];
    v = v > 0.f ? v : (__expf(v) - 1.f);     // ELU
    ob[idx] = to_bf16(v);
}

__global__ void fin2_k(float* __restrict__ out, const float* __restrict__ b2) {
    long idx = blockIdx.x * 256L + threadIdx.x;
    if (idx >= (long)N_NODES * HID) return;
    out[idx] += b2[(int)(idx & (HID - 1))];
    if (idx == 0) {                           // ptr = [0, N_USERS, N_NODES]
        out[(long)N_NODES * HID + 0] = 0.f;
        out[(long)N_NODES * HID + 1] = (float)N_USERS;
        out[(long)N_NODES * HID + 2] = (float)N_NODES;
    }
}

// ---------------- host launcher ----------------
static inline int cdiv_l(long a, long b) { return (int)((a + b - 1) / b); }

extern "C" void kernel_launch(void* const* d_in, const int* in_sizes, int n_in,
                              void* d_out, int out_size, void* d_ws, size_t ws_size,
                              hipStream_t stream) {
    (void)n_in; (void)out_size; (void)ws_size;
    const float* mov_x    = (const float*)d_in[2];
    const int*   eidx     = (const int*)  d_in[3];
    const float* user_emb = (const float*)d_in[4];
    const float* item_emb = (const float*)d_in[5];
    const float* gW1 = (const float*)d_in[6],  *gb1 = (const float*)d_in[7];
    const float* gW2 = (const float*)d_in[8],  *gb2 = (const float*)d_in[9];
    const float* tW1 = (const float*)d_in[10], *tb1 = (const float*)d_in[11];
    const float* tW2 = (const float*)d_in[12], *tb2 = (const float*)d_in[13];
    const float* Ws1 = (const float*)d_in[14], *Wd1 = (const float*)d_in[15];
    const float* as1 = (const float*)d_in[16], *ad1 = (const float*)d_in[17];
    const float* b1  = (const float*)d_in[18];
    const float* Ws2 = (const float*)d_in[19], *Wd2 = (const float*)d_in[20];
    const float* as2 = (const float*)d_in[21], *ad2 = (const float*)d_in[22];
    const float* b2  = (const float*)d_in[23];
    float* out = (float*)d_out;
    const int E_ = in_sizes[3] / 2;
    const int* src = eidx;
    const int* dst = eidx + E_;

    // ---- workspace layout (all sizes multiples of 256 bytes) ----
    char* ws = (char*)d_ws;
    size_t o = 0;
    __bf16* xb = (__bf16*)(ws + o); o += (size_t)N_NODES * DM * 2;    // 46.08 MB
    float*  hs = (float*) (ws + o); o += (size_t)N_NODES * HID * 4;   // 122.88 MB
    char*   accR = ws + o;
    float*  acc = (float*)accR;     o += (size_t)N_NODES * HID * 4;   // 122.88 MB
    __bf16* ob = (__bf16*)(ws + o); o += (size_t)N_NODES * HID * 2;   // 61.44 MB
    float*  als = (float*)(ws + o); o += (size_t)N_NODES * H1 * 4;
    float*  ald = (float*)(ws + o); o += (size_t)N_NODES * H1 * 4;
    unsigned* mmaxk = (unsigned*)(ws + o); o += (size_t)N_NODES * H1 * 4;
    float*  ssum = (float*)(ws + o); o += (size_t)N_NODES * H1 * 4;
    __bf16* gW1t = (__bf16*)(ws + o); o += (size_t)EMB * KP_G * 2;
    __bf16* gW2t = (__bf16*)(ws + o); o += (size_t)EMB * EMB * 2;
    __bf16* tW1t = (__bf16*)(ws + o); o += (size_t)EMB * KP_T * 2;
    __bf16* tW2t = (__bf16*)(ws + o); o += (size_t)EMB * EMB * 2;
    __bf16* Ws1t = (__bf16*)(ws + o); o += (size_t)HID * DM * 2;
    __bf16* Ws2t = (__bf16*)(ws + o); o += (size_t)HID * HID * 2;
    float* Wsa1 = (float*)(ws + o); o += (size_t)DM * H1 * 4;
    float* Wda1 = (float*)(ws + o); o += (size_t)DM * H1 * 4;
    float* Wsa2 = (float*)(ws + o); o += (size_t)HID * 4;
    float* Wda2 = (float*)(ws + o); o += (size_t)HID * 4;
    // feature-stage temporaries aliased into the acc region (dead before acc is zeroed)
    size_t fo = 0;
    __bf16* gfb  = (__bf16*)(accR + fo); fo += (size_t)N_ITEMS * KP_G * 2;
    __bf16* tfb  = (__bf16*)(accR + fo); fo += (size_t)N_ITEMS * KP_T * 2;
    float*  t1f  = (float*) (accR + fo); fo += (size_t)N_ITEMS * EMB * 4;
    __bf16* t1b  = (__bf16*)(accR + fo); fo += (size_t)N_ITEMS * EMB * 2;
    float*  gout = (float*) (accR + fo); fo += (size_t)N_ITEMS * EMB * 4;
    float*  tout = (float*) (accR + fo); fo += (size_t)N_ITEMS * EMB * 4;

    auto gemm_grid = [](int M, int N, int NT) {
        long waves = (long)(M / 16) * (N / (NT * 16));
        return (int)((waves + 7) / 8);       // 8 waves (256 threads) per block
    };

    // ---- weight prep ----
    cvtw_k<<<cdiv_l((long)EMB * KP_G, 256), 256, 0, stream>>>(gW1, gW1t, N_GENRES, EMB, KP_G);
    cvtw_k<<<cdiv_l((long)EMB * EMB, 256), 256, 0, stream>>>(gW2, gW2t, EMB, EMB, EMB);
    cvtw_k<<<cdiv_l((long)EMB * KP_T, 256), 256, 0, stream>>>(tW1, tW1t, N_TAGS, EMB, KP_T);
    cvtw_k<<<cdiv_l((long)EMB * EMB, 256), 256, 0, stream>>>(tW2, tW2t, EMB, EMB, EMB);
    cvtw_k<<<cdiv_l((long)HID * DM, 256), 256, 0, stream>>>(Ws1, Ws1t, DM, HID, DM);
    cvtw_k<<<cdiv_l((long)HID * HID, 256), 256, 0, stream>>>(Ws2, Ws2t, HID, HID, HID);
    fold_k<<<cdiv_l((long)DM * H1, 256), 256, 0, stream>>>(Ws1, as1, Wsa1, DM, HID, H1, C1V);
    fold_k<<<cdiv_l((long)DM * H1, 256), 256, 0, stream>>>(Wd1, ad1, Wda1, DM, HID, H1, C1V);
    fold_k<<<cdiv_l((long)HID, 256), 256, 0, stream>>>(Ws2, as2, Wsa2, HID, HID, 1, HID);
    fold_k<<<cdiv_l((long)HID, 256), 256, 0, stream>>>(Wd2, ad2, Wda2, HID, HID, 1, HID);

    // ---- movie feature MLPs (WMMA bf16) ----
    featcvt_k<<<cdiv_l((long)N_ITEMS * (KP_G + KP_T), 256), 256, 0, stream>>>(mov_x, gfb, tfb);
    wmma_gemm_k<4><<<gemm_grid(N_ITEMS, EMB, 4), 256, 0, stream>>>(gfb, gW1t, t1f, gb1, N_ITEMS, EMB, KP_G, 1);
    f2b_k<<<cdiv_l((long)N_ITEMS * EMB, 256), 256, 0, stream>>>(t1f, t1b, (long)N_ITEMS * EMB);
    wmma_gemm_k<4><<<gemm_grid(N_ITEMS, EMB, 4), 256, 0, stream>>>(t1b, gW2t, gout, gb2, N_ITEMS, EMB, EMB, 0);
    wmma_gemm_k<4><<<gemm_grid(N_ITEMS, EMB, 4), 256, 0, stream>>>(tfb, tW1t, t1f, tb1, N_ITEMS, EMB, KP_T, 1);
    f2b_k<<<cdiv_l((long)N_ITEMS * EMB, 256), 256, 0, stream>>>(t1f, t1b, (long)N_ITEMS * EMB);
    wmma_gemm_k<4><<<gemm_grid(N_ITEMS, EMB, 4), 256, 0, stream>>>(t1b, tW2t, tout, tb2, N_ITEMS, EMB, EMB, 0);

    // ---- node features x (bf16) ----
    build_x_k<<<cdiv_l((long)N_NODES * DM, 256), 256, 0, stream>>>(user_emb, item_emb, gout, tout, xb);

    // ---- GAT layer 1 ----
    wmma_gemm_k<8><<<gemm_grid(N_NODES, HID, 8), 256, 0, stream>>>(xb, Ws1t, hs, nullptr, N_NODES, HID, DM, 0);
    al_k<H1><<<cdiv_l(N_NODES, 256), 256, 0, stream>>>(xb, Wsa1, Wda1, als, ald, DM);
    fill_k<<<cdiv_l((long)N_NODES * HID, 256), 256, 0, stream>>>((unsigned*)acc, 0u, (long)N_NODES * HID);
    fill_k<<<cdiv_l((long)N_NODES * H1, 256), 256, 0, stream>>>(mmaxk, 0u, (long)N_NODES * H1);
    fill_k<<<cdiv_l((long)N_NODES * H1, 256), 256, 0, stream>>>((unsigned*)ssum, 0u, (long)N_NODES * H1);
    edge_max_k<<<cdiv_l((long)E_ * H1, 256), 256, 0, stream>>>(src, dst, als, ald, mmaxk, E_, 3);
    edge_sum_k<<<cdiv_l((long)E_ * H1, 256), 256, 0, stream>>>(src, dst, als, ald, mmaxk, ssum, E_, 3);
    edge_aggr_k<<<E_, 256, 0, stream>>>(src, dst, als, ald, mmaxk, ssum, hs, acc, E_, H1, 5);
    fin1_k<<<cdiv_l((long)N_NODES * HID, 256), 256, 0, stream>>>(acc, b1, ob);

    // ---- GAT layer 2 (H=1) ----
    wmma_gemm_k<8><<<gemm_grid(N_NODES, HID, 8), 256, 0, stream>>>(ob, Ws2t, hs, nullptr, N_NODES, HID, HID, 0);
    al_k<1><<<cdiv_l(N_NODES, 256), 256, 0, stream>>>(ob, Wsa2, Wda2, als, ald, HID);
    fill_k<<<cdiv_l((long)N_NODES * HID, 256), 256, 0, stream>>>((unsigned*)out, 0u, (long)N_NODES * HID);
    fill_k<<<cdiv_l((long)N_NODES, 256), 256, 0, stream>>>(mmaxk, 0u, (long)N_NODES);
    fill_k<<<cdiv_l((long)N_NODES, 256), 256, 0, stream>>>((unsigned*)ssum, 0u, (long)N_NODES);
    edge_max_k<<<cdiv_l((long)E_, 256), 256, 0, stream>>>(src, dst, als, ald, mmaxk, E_, 0);
    edge_sum_k<<<cdiv_l((long)E_, 256), 256, 0, stream>>>(src, dst, als, ald, mmaxk, ssum, E_, 0);
    edge_aggr_k<<<E_, 256, 0, stream>>>(src, dst, als, ald, mmaxk, ssum, hs, out, E_, 1, 8);
    fin2_k<<<cdiv_l((long)N_NODES * HID, 256), 256, 0, stream>>>(out, b2);
}
